// GRUGNNCell_21629455302676
// MI455X (gfx1250) — matrix operand order repaired and checked by reading, not compiled, not run on hardware
//
#include <hip/hip_runtime.h>
#include <hip/hip_bf16.h>

// B=512, N=128, F=256, H=256
typedef __attribute__((ext_vector_type(16))) __bf16 v16bf;
typedef __attribute__((ext_vector_type(8)))  __bf16 v8bf;
typedef __attribute__((ext_vector_type(4)))  __bf16 v4bf;
typedef __attribute__((ext_vector_type(8)))  float  v8f;
typedef __attribute__((ext_vector_type(4)))  unsigned int u32x4;
typedef __attribute__((ext_vector_type(8)))  int i32x8;
typedef __attribute__((ext_vector_type(4)))  int i32x4;

#define NB   512
#define NN   128
#define NF   256
#define NH   256
#define KP1  136   // kernel1 LDS row stride (bf16), 272B = 17*16B aligned
#define KP2  520   // kernel2 LDS row stride (bf16), 1040B = 65*16B aligned
                   // = 512 data + 8 pad bf16  -> pad 4 DWORDs every 256 DWORDs

// ---------------------------------------------------------------------------
// Kernel 0: build WT[gate][n][k] (bf16, column-major weights, K stacked 512)
//   WT[g][n][k] = (k<256 ? W{g}1[k][n] : W{g}2[k-256][n])
// ---------------------------------------------------------------------------
__global__ void wt_build_kernel(const float* __restrict__ Wz1, const float* __restrict__ Wz2,
                                const float* __restrict__ Wr1, const float* __restrict__ Wr2,
                                const float* __restrict__ Wh1, const float* __restrict__ Wh2,
                                __bf16* __restrict__ WT) {
  int idx = blockIdx.x * blockDim.x + threadIdx.x;   // 0 .. 3*256*512-1
  int k = idx & 511;
  int n = (idx >> 9) & 255;
  int g = idx >> 17;
  const float* W1 = (g == 0) ? Wz1 : (g == 1) ? Wr1 : Wh1;
  const float* W2 = (g == 0) ? Wz2 : (g == 1) ? Wr2 : Wh2;
  float v = (k < 256) ? W1[(size_t)k * NH + n] : W2[(size_t)(k - 256) * NH + n];
  WT[idx] = (__bf16)v;
}

// ---------------------------------------------------------------------------
// Kernel 1: AXH[b] = A[b] @ [X[b] | hidden[b]]   -> bf16 [128 x 512]
// One block per (batch, 128-col chunk). 256 threads = 8 waves.
// ---------------------------------------------------------------------------
__global__ void __launch_bounds__(256)
prop_kernel(const float* __restrict__ X, const float* __restrict__ A,
            const float* __restrict__ Hin, __bf16* __restrict__ AXH) {
  __shared__ __bf16 ldsA[128 * KP1];   // A[b], row-major [m][k]
  __shared__ __bf16 ldsB[128 * KP1];   // operand chunk, TRANSPOSED [n][k]

  const int b = blockIdx.x >> 2;
  const int c = blockIdx.x & 3;
  const int tid = threadIdx.x;

  const float* Ab = A + (size_t)b * NN * NN;
  const float* src;
  int src_col0;
  if (c < 2) { src = X   + (size_t)b * NN * NF; src_col0 = c * 128; }
  else       { src = Hin + (size_t)b * NN * NH; src_col0 = (c - 2) * 128; }
  const int out_col0 = c * 128;

  // Stage A -> ldsA (convert fp32 -> bf16), vectorized float4 reads.
  for (int i = tid; i < 128 * 32; i += 256) {       // 4096 float4
    int row = i >> 5, c4 = i & 31;
    float4 f = ((const float4*)Ab)[i];
    v4bf p; p[0] = (__bf16)f.x; p[1] = (__bf16)f.y; p[2] = (__bf16)f.z; p[3] = (__bf16)f.w;
    *(v4bf*)&ldsA[row * KP1 + c4 * 4] = p;          // ds_store_b64
  }
  // Stage operand chunk transposed: ldsB[n][k] = src[k][src_col0+n]
  for (int i = tid; i < 128 * 32; i += 256) {
    int row = i >> 5, c4 = i & 31;                  // row = k index
    float4 f = ((const float4*)(src + (size_t)row * 256 + src_col0))[c4];
    ldsB[(c4 * 4 + 0) * KP1 + row] = (__bf16)f.x;
    ldsB[(c4 * 4 + 1) * KP1 + row] = (__bf16)f.y;
    ldsB[(c4 * 4 + 2) * KP1 + row] = (__bf16)f.z;
    ldsB[(c4 * 4 + 3) * KP1 + row] = (__bf16)f.w;
  }
  __syncthreads();

  const int w = tid >> 5, lane = tid & 31;
  const int lhi = lane >> 4, llo = lane & 15;
  const int m0 = w * 16;

  v8f acc[8];
#pragma unroll
  for (int t = 0; t < 8; ++t) acc[t] = {};

#pragma unroll
  for (int ks = 0; ks < 4; ++ks) {                  // K = 128, 32 per step
    const int k0 = ks * 32;
    // A fragment (16x32 bf16): lane holds row m0+llo, K in [16*lhi, +16)
    v16bf afrag;
    {
      v8bf* ap = (v8bf*)&afrag;
      const __bf16* ab = &ldsA[(m0 + llo) * KP1 + k0 + lhi * 16];
      ap[0] = *(const v8bf*)ab;
      ap[1] = *(const v8bf*)(ab + 8);
    }
#pragma unroll
    for (int nt = 0; nt < 8; ++nt) {
      // B fragment (32x16 bf16): lane holds col nt*16+llo, K in [16*lhi, +16)
      v16bf bfrag;
      v8bf* bp = (v8bf*)&bfrag;
      const __bf16* bb = &ldsB[(nt * 16 + llo) * KP1 + k0 + lhi * 16];
      bp[0] = *(const v8bf*)bb;
      bp[1] = *(const v8bf*)(bb + 8);
      acc[nt] = __builtin_amdgcn_wmma_f32_16x16x32_bf16(
          false, afrag, false, bfrag, (short)0, acc[nt], false, false);
    }
  }

  // Store: C/D layout -> element (m = r + 8*lhi, n = llo) per VGPR r
  __bf16* ob = AXH + (size_t)b * NN * 512 + out_col0;
#pragma unroll
  for (int nt = 0; nt < 8; ++nt) {
    const int n_g = nt * 16 + llo;
#pragma unroll
    for (int r = 0; r < 8; ++r) {
      const int m_g = m0 + r + lhi * 8;
      ob[(size_t)m_g * 512 + n_g] = (__bf16)acc[nt][r];
    }
  }
}

// ---------------------------------------------------------------------------
// Kernel 2: per batch, gates + GRU blend.
//   pre_z = AXH @ [Wz1;Wz2] ; pre_r = AXH @ [Wr1;Wr2]
//   h1 = AX @ Wh1 ; h2 = AH @ Wh2   (split accumulators at K=256)
//   z = sigmoid(pre_z + bz); r = sigmoid(pre_r + br)
//   h = z*hidden + (1-z)*tanh(h1 + r*h2 + bh)
// ---------------------------------------------------------------------------
__global__ void __launch_bounds__(256)
gate_kernel(const __bf16* __restrict__ AXH, const __bf16* __restrict__ WT,
            const float* __restrict__ Hin,
            const float* __restrict__ pbz, const float* __restrict__ pbr,
            const float* __restrict__ pbh, float* __restrict__ out) {
  __shared__ __bf16 ldsX[128 * KP2];   // AXH[b], row-major [m][k], padded rows

  const int b = blockIdx.x;
  const int tid = threadIdx.x;
  const __bf16* src = AXH + (size_t)b * NN * 512;

#if __has_builtin(__builtin_amdgcn_tensor_load_to_lds) && __has_builtin(__builtin_amdgcn_s_wait_tensorcnt)
  // TDM staging: one DMA moves the whole 128x512 bf16 tile, inserting the
  // 4-DWORD LDS row pad every 256 DWORDs (matches KP2=520 layout).
  if (tid < 32) {                       // one wave issues; EXEC ignored by TDM
    unsigned lds_off = (unsigned)(uintptr_t)(void*)ldsX;  // low 32 bits = LDS byte addr
    unsigned long long ga = (unsigned long long)(uintptr_t)src;
    u32x4 g0;
    g0[0] = 1u;                                         // count=1, user D#
    g0[1] = lds_off;                                    // lds_addr (bytes)
    g0[2] = (unsigned)(ga & 0xffffffffu);               // global_addr[31:0]
    g0[3] = (unsigned)((ga >> 32) & 0x01ffffffu)        // global_addr[56:32]
          | (2u << 30);                                 // type = 2 ("image")
    i32x8 g1;
    g1[0] = (1 << 16)   // data_size = 2 bytes (bf16)
          | (1 << 20)   // pad_enable
          | (7 << 22)   // pad_interval: every 256 DWORDs (one 512-bf16 row)
          | (3 << 25);  // pad_amount: 4 DWORDs (8 bf16) -> row stride 520
    g1[1] = (int)(512u << 16);   // tensor_dim0 = 512 (low16; hi16 in g1[2][15:0]=0)
    g1[2] = (int)(128u << 16);   // tensor_dim1 = 128
    g1[3] = (int)(512u << 16);   // tile_dim0 = 512
    g1[4] = 128;                 // tile_dim1 = 128 ; tile_dim2 = 0
    g1[5] = 512;                 // tensor_dim0_stride = 512 elements
    g1[6] = 0;
    g1[7] = 0;
    i32x4 gz4 = {0, 0, 0, 0};            // groups 2/3 unused (2D tensor)
    i32x8 gz8 = {0, 0, 0, 0, 0, 0, 0, 0};
    // 6-arg form (amdgpu-toolchain clang-23)
    __builtin_amdgcn_tensor_load_to_lds(g0, g1, gz4, gz4, gz8, 0);
    __builtin_amdgcn_s_wait_tensorcnt(0);
  }
#else
  // Fallback: VGPR round-trip staging, 8192 x 16B
  for (int i = tid; i < 8192; i += 256) {
    int row = i >> 6, c8 = i & 63;
    *(v8bf*)&ldsX[row * KP2 + c8 * 8] = *(const v8bf*)&src[(size_t)row * 512 + c8 * 8];
  }
#endif
  // Warm weights toward caches while staging completes
  __builtin_prefetch(WT + (size_t)tid * 64, 0, 1);
  __syncthreads();

  const int w = tid >> 5, lane = tid & 31;
  const int lhi = lane >> 4, llo = lane & 15;
  const int m0 = w * 16;

  const __bf16* WTz = WT;
  const __bf16* WTr = WT + (size_t)256 * 512;
  const __bf16* WTh = WT + (size_t)2 * 256 * 512;

  const float* hb = Hin + (size_t)b * NN * NH;
  float*       ob = out + (size_t)b * NN * NH;

#pragma unroll 1
  for (int nt = 0; nt < 16; ++nt) {
    const int n0 = nt * 16;
    v8f az = {}, ar = {}, ah1 = {}, ah2 = {};
    const size_t wrow = ((size_t)(n0 + llo)) * 512 + lhi * 16;
    const __bf16* wz = WTz + wrow;
    const __bf16* wr = WTr + wrow;
    const __bf16* wh = WTh + wrow;

#pragma unroll 2
    for (int ks = 0; ks < 16; ++ks) {               // K = 512
      const int k0 = ks * 32;
      v16bf afrag;
      {
        v8bf* ap = (v8bf*)&afrag;
        const __bf16* ab = &ldsX[(m0 + llo) * KP2 + k0 + lhi * 16];
        ap[0] = *(const v8bf*)ab;
        ap[1] = *(const v8bf*)(ab + 8);
      }
      v16bf bz_, br_, bh_;
      { v8bf* p = (v8bf*)&bz_; p[0] = *(const v8bf*)(wz + k0); p[1] = *(const v8bf*)(wz + k0 + 8); }
      { v8bf* p = (v8bf*)&br_; p[0] = *(const v8bf*)(wr + k0); p[1] = *(const v8bf*)(wr + k0 + 8); }
      { v8bf* p = (v8bf*)&bh_; p[0] = *(const v8bf*)(wh + k0); p[1] = *(const v8bf*)(wh + k0 + 8); }

      az = __builtin_amdgcn_wmma_f32_16x16x32_bf16(false, afrag, false, bz_, (short)0, az, false, false);
      ar = __builtin_amdgcn_wmma_f32_16x16x32_bf16(false, afrag, false, br_, (short)0, ar, false, false);
      if (ks < 8)  // uniform branch: EXEC stays all-ones for WMMA
        ah1 = __builtin_amdgcn_wmma_f32_16x16x32_bf16(false, afrag, false, bh_, (short)0, ah1, false, false);
      else
        ah2 = __builtin_amdgcn_wmma_f32_16x16x32_bf16(false, afrag, false, bh_, (short)0, ah2, false, false);
    }

    // Epilogue: element (m = r + 8*lhi, n = llo) of this 16x16 tile
    const int n_g = n0 + llo;
#pragma unroll
    for (int r = 0; r < 8; ++r) {
      const int m_g = m0 + r + lhi * 8;
      const size_t boff = (size_t)m_g * NH + n_g;   // bias index [N,H]
      float z  = 1.0f / (1.0f + __expf(-(az[r] + pbz[boff])));
      float rr = 1.0f / (1.0f + __expf(-(ar[r] + pbr[boff])));
      float hv = tanhf(ah1[r] + rr * ah2[r] + pbh[boff]);
      float hp = hb[boff];
      ob[boff] = z * hp + (1.0f - z) * hv;
    }
  }
}

// ---------------------------------------------------------------------------
extern "C" void kernel_launch(void* const* d_in, const int* in_sizes, int n_in,
                              void* d_out, int out_size, void* d_ws, size_t ws_size,
                              hipStream_t stream) {
  const float* X   = (const float*)d_in[0];
  const float* A   = (const float*)d_in[1];
  const float* Hin = (const float*)d_in[2];
  const float* Wz1 = (const float*)d_in[3];
  const float* Wz2 = (const float*)d_in[4];
  const float* Wr1 = (const float*)d_in[5];
  const float* Wr2 = (const float*)d_in[6];
  const float* Wh1 = (const float*)d_in[7];
  const float* Wh2 = (const float*)d_in[8];
  const float* bz  = (const float*)d_in[9];
  const float* br  = (const float*)d_in[10];
  const float* bh  = (const float*)d_in[11];
  float* out = (float*)d_out;

  // Workspace layout: [0, 768KB) WT bf16 ; [1MB, 1MB+64MB) AXH bf16
  __bf16* WT  = (__bf16*)d_ws;
  __bf16* AXH = (__bf16*)((char*)d_ws + (1u << 20));

  wt_build_kernel<<<(3 * 256 * 512) / 256, 256, 0, stream>>>(Wz1, Wz2, Wr1, Wr2, Wh1, Wh2, WT);
  prop_kernel<<<NB * 4, 256, 0, stream>>>(X, A, Hin, AXH);
  gate_kernel<<<NB, 256, 0, stream>>>(AXH, WT, Hin, bz, br, bh, out);
}
